// LabelRestrictedSelfAttention_89163521065537
// MI455X (gfx1250) — matrix operand order repaired and checked by reading, not compiled, not run on hardware
//
#include <hip/hip_runtime.h>

// ---------------------------------------------------------------------------
// Label-restricted self-attention for MI455X (gfx1250, wave32, WMMA)
//   S = Q K^T  (2048x2048x4096, bf16 WMMA, f32 acc)  -> masked -> softmax
//   O = P V    (2048x4096x2048, bf16 WMMA, f32 acc)  -> validity zeroing
// Q/K/V rows are computed on the fly from x: row r of the flat qkv matrix is
//   x[b*256 + (r%768)/3, :] * w[r%768] + bias[r%768],  b = r/768
// with q: r=i, k: r=2048+i, v: r=4096+i.
// Workspace: S (2048*2048 f32 = 16.8 MB), L2-resident.
// ---------------------------------------------------------------------------

typedef __attribute__((ext_vector_type(16))) __bf16 v16bf;
typedef __attribute__((ext_vector_type(8)))  float  v8f;

#define N_TOK 2048
#define DIM   4096
#define BM 128
#define BN 128
#define BK 64
#define LDA 72   // bf16 elements per LDS row (64 + 8 pad -> bank spread)
#define LDB 72
#define NEG_BIG (-3.4028234663852886e38f)

__device__ __forceinline__ __bf16 to_bf16(float f) {
  unsigned u = __builtin_bit_cast(unsigned, f);
  unsigned r = u + 0x7FFFu + ((u >> 16) & 1u);   // round-to-nearest-even
  unsigned short h = (unsigned short)(r >> 16);
  return __builtin_bit_cast(__bf16, h);
}

// Map a flat qkv row r (0..6143) to its x row and per-row scale/offset.
__device__ __forceinline__ void qkv_row(int r, const float* __restrict__ w,
                                        const float* __restrict__ bias,
                                        int& xrow, float& sc, float& of) {
  int bb  = r / 768;
  int rem = r - bb * 768;
  xrow = bb * 256 + rem / 3;
  sc = w[rem];
  of = bias[rem];
}

// 8 waves arranged 4 (M) x 2 (N); each wave computes a 32x64 subtile as
// 2x4 fragments of 16x16, looping kk over the 64-wide LDS K-chunk.
// A in LDS: [m][k] row-major (LDA).  B in LDS: [n][k] row-major (LDB).
__device__ __forceinline__ void mma_tile(const __bf16* __restrict__ As,
                                         const __bf16* __restrict__ Bs,
                                         v8f acc[2][4], int wvM, int wvN, int lane) {
  const int l15 = lane & 15;
  const int hi  = lane >> 4;
#pragma unroll
  for (int kk = 0; kk < BK; kk += 32) {
    v16bf a[2];
#pragma unroll
    for (int mi = 0; mi < 2; ++mi) {
      const int m = wvM * 32 + mi * 16 + l15;
      const int khA = hi * 8;
#pragma unroll
      for (int j = 0; j < 8; ++j) {
        const int k0 = kk + khA + ((j < 4) ? (2 * j) : (2 * j + 8));
        a[mi][2 * j]     = As[m * LDA + k0];
        a[mi][2 * j + 1] = As[m * LDA + k0 + 1];
      }
    }
#pragma unroll
    for (int ni = 0; ni < 4; ++ni) {
      const int n = wvN * 64 + ni * 16 + l15;
      const int khB = hi * 8;
      v16bf b;
#pragma unroll
      for (int j = 0; j < 8; ++j) {
        const int k0 = kk + khB + ((j < 4) ? (2 * j) : (2 * j + 8));
        b[2 * j]     = Bs[n * LDB + k0];
        b[2 * j + 1] = Bs[n * LDB + k0 + 1];
      }
      acc[0][ni] = __builtin_amdgcn_wmma_f32_16x16x32_bf16(
          false, a[0], false, b, (short)0, acc[0][ni], false, false);
      acc[1][ni] = __builtin_amdgcn_wmma_f32_16x16x32_bf16(
          false, a[1], false, b, (short)0, acc[1][ni], false, false);
    }
  }
}

// ---------------------------------------------------------------------------
// Kernel 1: S = Q K^T with label mask fused into the epilogue.
// ---------------------------------------------------------------------------
__global__ __launch_bounds__(256) void qk_gemm_kernel(
    const float* __restrict__ x, const float* __restrict__ w,
    const float* __restrict__ bias, const int* __restrict__ labels,
    float* __restrict__ S) {
  __shared__ __bf16 As[BM * LDA];
  __shared__ __bf16 Bs[BN * LDB];

  const int t = threadIdx.x;
  const int lane = t & 31, wv = t >> 5;
  const int wvM = wv & 3, wvN = wv >> 2;
  const int i0 = blockIdx.y * BM;   // query rows
  const int j0 = blockIdx.x * BN;   // key rows

  v8f acc[2][4];
#pragma unroll
  for (int mi = 0; mi < 2; ++mi)
#pragma unroll
    for (int ni = 0; ni < 4; ++ni)
#pragma unroll
      for (int e = 0; e < 8; ++e) acc[mi][ni][e] = 0.f;

  for (int d0 = 0; d0 < DIM; d0 += BK) {
    // A tile: Q rows i0..i0+127, cols d0..d0+63 (scale/bias fused, -> bf16)
#pragma unroll
    for (int p = 0; p < 8; ++p) {
      const int ml = (t >> 4) + p * 16;
      const int c4 = (t & 15) * 4;
      int xrow; float sc, of;
      qkv_row(i0 + ml, w, bias, xrow, sc, of);
      const float4 v = *(const float4*)(x + (size_t)xrow * DIM + d0 + c4);
      __bf16* dst = &As[ml * LDA + c4];
      dst[0] = to_bf16(v.x * sc + of); dst[1] = to_bf16(v.y * sc + of);
      dst[2] = to_bf16(v.z * sc + of); dst[3] = to_bf16(v.w * sc + of);
    }
    // B tile: K rows j0..j0+127, cols d0..d0+63 stored [n][k]
#pragma unroll
    for (int p = 0; p < 8; ++p) {
      const int nl = (t >> 4) + p * 16;
      const int c4 = (t & 15) * 4;
      int xrow; float sc, of;
      qkv_row(2048 + j0 + nl, w, bias, xrow, sc, of);
      const float4 v = *(const float4*)(x + (size_t)xrow * DIM + d0 + c4);
      __bf16* dst = &Bs[nl * LDB + c4];
      dst[0] = to_bf16(v.x * sc + of); dst[1] = to_bf16(v.y * sc + of);
      dst[2] = to_bf16(v.z * sc + of); dst[3] = to_bf16(v.w * sc + of);
    }
    __syncthreads();
    mma_tile(As, Bs, acc, wvM, wvN, lane);
    __syncthreads();
  }

  // Epilogue: label mask, write scores.
  const int l15 = lane & 15, hi = lane >> 4;
#pragma unroll
  for (int mi = 0; mi < 2; ++mi) {
#pragma unroll
    for (int ni = 0; ni < 4; ++ni) {
      const int gn = j0 + wvN * 64 + ni * 16 + l15;
      const int lj = labels[gn];
#pragma unroll
      for (int r = 0; r < 8; ++r) {
        const int gm = i0 + wvM * 32 + mi * 16 + r + 8 * hi;
        const int li = labels[gm];
        const bool keep = (li == lj) && (li >= 0);
        S[(size_t)gm * N_TOK + gn] = keep ? acc[mi][ni][r] : NEG_BIG;
      }
    }
  }
}

// ---------------------------------------------------------------------------
// Kernel 2: in-place row softmax over S (one block per row of 2048).
// ---------------------------------------------------------------------------
__global__ __launch_bounds__(256) void softmax_kernel(float* __restrict__ S) {
  const int row = blockIdx.x;
  float* p = S + (size_t)row * N_TOK;
  const int t = threadIdx.x;
  __shared__ float red[256];

  float v[8];
  float m = NEG_BIG;
#pragma unroll
  for (int e = 0; e < 8; ++e) { v[e] = p[t + e * 256]; m = fmaxf(m, v[e]); }
  red[t] = m;
  __syncthreads();
  for (int off = 128; off > 0; off >>= 1) {
    if (t < off) red[t] = fmaxf(red[t], red[t + off]);
    __syncthreads();
  }
  const float rmax = red[0];
  __syncthreads();

  float s = 0.f;
#pragma unroll
  for (int e = 0; e < 8; ++e) { v[e] = __expf(v[e] - rmax); s += v[e]; }
  red[t] = s;
  __syncthreads();
  for (int off = 128; off > 0; off >>= 1) {
    if (t < off) red[t] += red[t + off];
    __syncthreads();
  }
  const float inv = 1.f / red[0];
#pragma unroll
  for (int e = 0; e < 8; ++e) p[t + e * 256] = v[e] * inv;
}

// ---------------------------------------------------------------------------
// Kernel 3: O = P V, with validity zeroing fused into the epilogue.
// ---------------------------------------------------------------------------
__global__ __launch_bounds__(256) void pv_gemm_kernel(
    const float* __restrict__ x, const float* __restrict__ w,
    const float* __restrict__ bias, const int* __restrict__ labels,
    const float* __restrict__ P, float* __restrict__ out) {
  __shared__ __bf16 As[BM * LDA];
  __shared__ __bf16 Bs[BN * LDB];

  const int t = threadIdx.x;
  const int lane = t & 31, wv = t >> 5;
  const int wvM = wv & 3, wvN = wv >> 2;
  const int i0 = blockIdx.y * BM;   // token rows
  const int d0 = blockIdx.x * BN;   // output dim cols

  v8f acc[2][4];
#pragma unroll
  for (int mi = 0; mi < 2; ++mi)
#pragma unroll
    for (int ni = 0; ni < 4; ++ni)
#pragma unroll
      for (int e = 0; e < 8; ++e) acc[mi][ni][e] = 0.f;

  for (int k0 = 0; k0 < N_TOK; k0 += BK) {
    // A tile: normalized P[i0+ml][k0..k0+63] f32 -> bf16
#pragma unroll
    for (int p = 0; p < 8; ++p) {
      const int ml = (t >> 4) + p * 16;
      const int c4 = (t & 15) * 4;
      const float4 v = *(const float4*)(P + (size_t)(i0 + ml) * N_TOK + k0 + c4);
      __bf16* dst = &As[ml * LDA + c4];
      dst[0] = to_bf16(v.x); dst[1] = to_bf16(v.y);
      dst[2] = to_bf16(v.z); dst[3] = to_bf16(v.w);
    }
    // B tile: V rows k0..k0+63, cols d0..d0+127, stored transposed [d][j]
#pragma unroll
    for (int p = 0; p < 8; ++p) {
      const int jl = (t >> 5) + p * 8;
      const int c4 = (t & 31) * 4;
      int xrow; float sc, of;
      qkv_row(4096 + k0 + jl, w, bias, xrow, sc, of);
      const float4 v = *(const float4*)(x + (size_t)xrow * DIM + d0 + c4);
      Bs[(c4 + 0) * LDB + jl] = to_bf16(v.x * sc + of);
      Bs[(c4 + 1) * LDB + jl] = to_bf16(v.y * sc + of);
      Bs[(c4 + 2) * LDB + jl] = to_bf16(v.z * sc + of);
      Bs[(c4 + 3) * LDB + jl] = to_bf16(v.w * sc + of);
    }
    __syncthreads();
    mma_tile(As, Bs, acc, wvM, wvN, lane);
    __syncthreads();
  }

  const int l15 = lane & 15, hi = lane >> 4;
#pragma unroll
  for (int mi = 0; mi < 2; ++mi) {
#pragma unroll
    for (int ni = 0; ni < 4; ++ni) {
      const int gn = d0 + wvN * 64 + ni * 16 + l15;
#pragma unroll
      for (int r = 0; r < 8; ++r) {
        const int gm = i0 + wvM * 32 + mi * 16 + r + 8 * hi;
        const float val = acc[mi][ni][r];
        out[(size_t)gm * DIM + gn] = (labels[gm] >= 0) ? val : 0.f;
      }
    }
  }
}

// ---------------------------------------------------------------------------
extern "C" void kernel_launch(void* const* d_in, const int* in_sizes, int n_in,
                              void* d_out, int out_size, void* d_ws, size_t ws_size,
                              hipStream_t stream) {
  (void)in_sizes; (void)n_in; (void)out_size; (void)ws_size;
  const float* x      = (const float*)d_in[0];
  const int*   labels = (const int*)d_in[1];
  const float* w      = (const float*)d_in[2];
  const float* b      = (const float*)d_in[3];
  float* out = (float*)d_out;
  float* S   = (float*)d_ws;   // 2048*2048*4 = 16.8 MB scratch

  dim3 blk(256);
  qk_gemm_kernel<<<dim3(N_TOK / BN, N_TOK / BM), blk, 0, stream>>>(x, w, b, labels, S);
  softmax_kernel<<<dim3(N_TOK), blk, 0, stream>>>(S);
  pv_gemm_kernel<<<dim3(DIM / BN, N_TOK / BM), blk, 0, stream>>>(x, w, b, labels, S, out);
}
